// Model_6210522710668
// MI455X (gfx1250) — matrix operand order repaired and checked by reading, not compiled, not run on hardware
//
#include <hip/hip_runtime.h>
#include <hip/hip_bf16.h>
#include <math.h>
#include <stdint.h>

typedef __attribute__((ext_vector_type(16))) __bf16 v16bf;
typedef __attribute__((ext_vector_type(8)))  float  v8f;
typedef __attribute__((ext_vector_type(4)))  unsigned int u32x4;
typedef __attribute__((ext_vector_type(8)))  int i32x8;
typedef __attribute__((ext_vector_type(4)))  int i32x4;

#define BATCH 65536
#define LSEQ  512
#define PDIM  96
#define NEXP  14

// ---------------- workspace layout (bytes) ----------------
// mu      : B*4            @ 0
// sd      : B*4            @ 262144
// gates   : 2B*4           @ 524288
// topi    : 2B*4           @ 1048576
// counts  : 16*4           @ 1572864
// lists   : 14*B*4         @ 1572928
// whi     : 14*96*512*2    @ 5242944
// wlo     : 14*96*512*2    @ 6619200
// w1      : 14*96*4        @ 7995456
// yk      : B*2*96*4       @ 8000832   (total ~55.6 MB)

__global__ void zero_counts_kernel(int* counts) {
    if (threadIdx.x < 16) counts[threadIdx.x] = 0;
}

// split Wr (f32) into bf16 hi/lo halves for the 3-product WMMA emulation
__global__ void wprep_kernel(const float* __restrict__ Wr,
                             __bf16* __restrict__ whi, __bf16* __restrict__ wlo, int n) {
    int i = blockIdx.x * blockDim.x + threadIdx.x;
    if (i >= n) return;
    float w = Wr[i];
    __bf16 h = (__bf16)w;
    whi[i] = h;
    wlo[i] = (__bf16)(w - (float)h);
}

// per-(e,p) row sums of Wr (for folding the normalization into an affine term)
__global__ void w1_kernel(const float* __restrict__ Wr, float* __restrict__ w1) {
    int i = blockIdx.x * blockDim.x + threadIdx.x;
    if (i >= NEXP * PDIM) return;
    const float* p = Wr + (size_t)i * LSEQ;
    float s = 0.f;
    for (int j = 0; j < LSEQ; ++j) s += p[j];
    w1[i] = s;
}

// One block (256 thr) per row: mean/var, LDS radix-2 FFT-512, periodogram,
// gate matvec, top-2 softmax, append row to per-expert gather lists.
__global__ __launch_bounds__(256)
void gate_kernel(const float* __restrict__ x,
                 const float* __restrict__ Wg, const float* __restrict__ bg,
                 float* __restrict__ mu_o, float* __restrict__ sd_o,
                 float* __restrict__ gates_o, int* __restrict__ topi_o,
                 int* __restrict__ counts, int* __restrict__ lists) {
    __shared__ float re[512];
    __shared__ float im[512];
    __shared__ float sred[16];
    __shared__ float gsh[16];
    const int b = blockIdx.x;
    const int t = threadIdx.x;
    const int wid = t >> 5, lid = t & 31;
    const float* xr = x + (size_t)b * LSEQ;

    float v0 = xr[t], v1 = xr[t + 256];
    float s = v0 + v1, q = v0 * v0 + v1 * v1;
    for (int off = 16; off > 0; off >>= 1) {
        s += __shfl_xor(s, off, 32);
        q += __shfl_xor(q, off, 32);
    }
    if (lid == 0) { sred[wid] = s; sred[wid + 8] = q; }
    __syncthreads();
    if (t == 0) {
        float ts = 0.f, tq = 0.f;
        for (int i = 0; i < 8; ++i) { ts += sred[i]; tq += sred[i + 8]; }
        float mu = ts * (1.0f / LSEQ);
        float var = tq * (1.0f / LSEQ) - mu * mu;
        if (var < 0.f) var = 0.f;
        mu_o[b] = mu;
        sd_o[b] = sqrtf(var + 1e-5f);
        sred[0] = mu;
    }
    __syncthreads();
    const float mu = sred[0];

    // bit-reversed load (9-bit indices)
    {
        int r0 = __brev(t) >> 23, r1 = __brev(t + 256) >> 23;
        re[r0] = v0 - mu; im[r0] = 0.f;
        re[r1] = v1 - mu; im[r1] = 0.f;
    }
    __syncthreads();

    // 9 radix-2 DIT stages, one butterfly per thread per stage
    for (int st = 1; st <= 9; ++st) {
        const int m = 1 << st, hm = m >> 1;
        const int grp = t / hm, pos = t % hm;
        const int i0 = grp * m + pos, i1 = i0 + hm;
        float ang = -6.28318530717958647692f * (float)pos / (float)m;
        float sw, cw; __sincosf(ang, &sw, &cw);
        float ur = re[i0], ui = im[i0];
        float vr = re[i1], vi = im[i1];
        float tr = vr * cw - vi * sw;
        float ti = vr * sw + vi * cw;
        re[i0] = ur + tr; im[i0] = ui + ti;
        re[i1] = ur - tr; im[i1] = ui - ti;
        __syncthreads();
    }

    // periodogram (first 256 bins), normalize, gate matvec
    float I = (re[t] * re[t] + im[t] * im[t]) * (1.0f / 512.0f);
    float si = I;
    for (int off = 16; off > 0; off >>= 1) si += __shfl_xor(si, off, 32);
    if (lid == 0) sred[wid] = si;
    __syncthreads();
    if (t == 0) {
        float ts = 0.f;
        for (int i = 0; i < 8; ++i) ts += sred[i];
        if (ts == 0.f) ts = 1.f;
        sred[0] = ts;
    }
    __syncthreads();
    const float inv_s = 1.0f / sred[0];
    re[t] = I;                       // reuse re[] as I storage
    __syncthreads();
    if (t < 16) {
        const float* wrow = Wg + t * 256;
        float acc = 0.f;
        for (int j = 0; j < 256; ++j) acc += re[j] * wrow[j];
        gsh[t] = (acc * inv_s + bg[t]) * (1.0f / 0.07f);
    }
    __syncthreads();
    if (t == 0) {
        float best = -1e30f, second = -1e30f;
        int bi = 0, si2 = 0;
        for (int e = 0; e < 16; ++e) {
            float g = gsh[e];
            if (g > best)        { second = best; si2 = bi; best = g; bi = e; }
            else if (g > second) { second = g; si2 = e; }
        }
        float g0 = 1.0f / (1.0f + __expf(second - best));
        gates_o[2 * b] = g0;
        gates_o[2 * b + 1] = 1.0f - g0;
        topi_o[2 * b] = bi;
        topi_o[2 * b + 1] = si2;
        if (bi < NEXP)  { int p = atomicAdd(&counts[bi], 1);  lists[bi * BATCH + p]  = (b << 1); }
        if (si2 < NEXP) { int p = atomicAdd(&counts[si2], 1); lists[si2 * BATCH + p] = (b << 1) | 1; }
    }
}

// fill yk slots owned by the trivial experts (mean=14, naive=15)
__global__ void init_yk_kernel(const float* __restrict__ x, const float* __restrict__ mu,
                               const int* __restrict__ topi, const float* __restrict__ gates,
                               float* __restrict__ yk, int total) {
    int idx = blockIdx.x * blockDim.x + threadIdx.x;
    if (idx >= total) return;
    int p = idx % PDIM;
    int bk = idx / PDIM;             // b*2 + k
    int e = topi[bk];
    if (e < NEXP) return;            // linear experts handled by WMMA kernel
    int b = bk >> 1;
    float v = (e == 14) ? mu[b] : x[(size_t)b * LSEQ + (LSEQ - 1)];
    yk[(size_t)bk * PDIM + p] = gates[bk] * v;
}

// One block per (expert, 16-row tile): Y[16,96] = X[16,512] @ W[e]^T via bf16
// WMMA with 3-term precision split, f32 accumulation. 6 waves, one 16x16
// N-tile per wave. X rows gathered into LDS by the Tensor Data Mover in
// gather mode (16-bit row indices, padded LDS rows); manual fallback keeps
// the identical LDS layout:  addr(r,k) = r*520 + k + 4*(k>>8)  floats.
__global__ __launch_bounds__(192)
void expert_kernel(const float* __restrict__ x,
                   const __bf16* __restrict__ whi, const __bf16* __restrict__ wlo,
                   const float* __restrict__ w1, const float* __restrict__ br,
                   const float* __restrict__ mu, const float* __restrict__ sd,
                   const float* __restrict__ gates,
                   const int* __restrict__ counts, const int* __restrict__ lists,
                   float* __restrict__ yk) {
    const int e = blockIdx.x;
    const int tile = blockIdx.y;
    const int cnt = counts[e];
    if (tile * 16 >= cnt) return;    // uniform early exit (before any WMMA)

    __shared__ __align__(16) float Xs[16 * 520];
    __shared__ int ent[16];
    const int tid = threadIdx.x;
    if (tid < 16) {
        int idx = tile * 16 + tid;
        ent[tid] = lists[e * BATCH + (idx < cnt ? idx : tile * 16)];  // clamp pad rows
    }
    __syncthreads();

#if defined(__has_builtin) && __has_builtin(__builtin_amdgcn_tensor_load_to_lds) && __has_builtin(__builtin_amdgcn_s_wait_tensorcnt)
    if (tid < 32) {                  // wave 0 issues the TDM gather
        unsigned idxpk[8];
#pragma unroll
        for (int i = 0; i < 8; ++i)
            idxpk[i] = ((unsigned)(ent[2 * i] >> 1) & 0xFFFFu) |
                       (((unsigned)(ent[2 * i + 1] >> 1) & 0xFFFFu) << 16);
        unsigned long long ga = (unsigned long long)(uintptr_t)x;
        unsigned lds_base = (unsigned)(uintptr_t)Xs;
        u32x4 g0; i32x8 g1; i32x4 g2, g3;
        // group0: count=1, gather_mode=1 (bit31), 16-bit indices (bit30=0)
        g0.x = 0x80000001u;
        g0.y = lds_base;                                   // lds_addr
        g0.z = (unsigned)ga;                               // global_addr[31:0]
        g0.w = (unsigned)((ga >> 32) & 0x1FFFFFFu) | (2u << 30);  // addr[56:32], type=2
        // group1: data_size=4B(code 2), pad_enable, pad_interval=256dw(7), pad_amount=4dw(3)
        g1[0] = (2 << 16) | (1 << 20) | (7 << 22) | (3 << 25);
        g1[1] = (int)(512u << 16);        // tensor_dim0 = 512 (low16 @ [63:48])
        g1[2] = 0;                        // tensor_dim0 hi=0, tensor_dim1 lo16 = 0
        g1[3] = (int)(1u | (512u << 16)); // tensor_dim1 = 65536 (hi16=1), tile_dim0 = 512
        g1[4] = 16;                       // tile_dim1 = #valid gather indices
        g1[5] = 512;                      // tensor_dim0_stride (data_size units)
        g1[6] = 0;
        g1[7] = 0;
        g2[0] = (int)idxpk[0]; g2[1] = (int)idxpk[1]; g2[2] = (int)idxpk[2]; g2[3] = (int)idxpk[3];
        g3[0] = (int)idxpk[4]; g3[1] = (int)idxpk[5]; g3[2] = (int)idxpk[6]; g3[3] = (int)idxpk[7];
        i32x8 gext = {0, 0, 0, 0, 0, 0, 0, 0};  // clang-23 6-arg form: extra operand
        __builtin_amdgcn_tensor_load_to_lds(g0, g1, g2, g3, gext, 0);
        __builtin_amdgcn_s_wait_tensorcnt(0);
    }
#else
    for (int i = tid; i < 16 * 128; i += 192) {
        int r = i >> 7, k = (i & 127) * 4;
        float4 v = *(const float4*)(x + ((size_t)(ent[r] >> 1)) * LSEQ + k);
        *(float4*)(&Xs[r * 520 + k + 4 * (k >> 8)]) = v;
    }
#endif
    __syncthreads();

    const int wave = tid / 32;       // n-tile index 0..5
    const int lane = tid & 31;
    const int nl   = lane & 15;
    const int half = lane >> 4;
    const int p    = wave * 16 + nl;
    const __bf16* wh = whi + ((size_t)e * PDIM + p) * LSEQ;
    const __bf16* wl = wlo + ((size_t)e * PDIM + p) * LSEQ;
    const float* arow = &Xs[(lane & 15) * 520];    // A: lane -> row M = lane%16

    v8f acc = {};
#pragma unroll 2
    for (int k0 = 0; k0 < LSEQ; k0 += 32) {
        __builtin_prefetch(wh + k0 + 128, 0, 1);   // global_prefetch of upcoming B data
        // A fragment (contiguous K, matching B): elems 0..7 -> K=k0+half*16+j,
        //                                     elems 8..15 -> K=k0+half*16+8+j
        const int ka = k0 + half * 16;
        const int kb = k0 + half * 16 + 8;
        const float* pa = arow + ka + 4 * (ka >> 8);   // padded-row address fn
        const float* pb = arow + kb + 4 * (kb >> 8);
        v16bf ahi, alo;
#pragma unroll
        for (int j = 0; j < 8; ++j) {
            float f0 = pa[j], f1 = pb[j];
            __bf16 h0 = (__bf16)f0, h1 = (__bf16)f1;
            ahi[j] = h0;  ahi[j + 8] = h1;
            alo[j] = (__bf16)(f0 - (float)h0);
            alo[j + 8] = (__bf16)(f1 - (float)h1);
        }
        // B fragment in contiguous K order (same as A): lane (n,half), elem i -> K=half*16+i
        v16bf bhi = *(const v16bf*)(wh + k0 + half * 16);
        v16bf blo = *(const v16bf*)(wl + k0 + half * 16);
        acc = __builtin_amdgcn_wmma_f32_16x16x32_bf16(false, ahi, false, bhi, (short)0, acc, false, false);
        acc = __builtin_amdgcn_wmma_f32_16x16x32_bf16(false, ahi, false, blo, (short)0, acc, false, false);
        acc = __builtin_amdgcn_wmma_f32_16x16x32_bf16(false, alo, false, bhi, (short)0, acc, false, false);
    }

    // eo = x·W + mu*(1-sum(W)) + sd*br ; write g*eo to the (b,slot) slot
    const float c_w1 = 1.0f - w1[e * PDIM + p];
    const float c_br = br[e * PDIM + p];
#pragma unroll
    for (int r = 0; r < 8; ++r) {
        int m = r + 8 * half;        // C/D layout: VGPR r holds M=r (+8 for upper half)
        if (tile * 16 + m < cnt) {
            int en = ent[m];         // en = b*2 + k
            int b = en >> 1;
            float val = acc[r] + mu[b] * c_w1 + sd[b] * c_br;
            yk[(size_t)en * PDIM + p] = gates[en] * val;
        }
    }
}

__global__ void combine_kernel(const float* __restrict__ yk, float* __restrict__ out, int total) {
    int i = blockIdx.x * blockDim.x + threadIdx.x;
    if (i >= total) return;
    int p = i % PDIM, b = i / PDIM;
    out[i] = yk[(size_t)b * (2 * PDIM) + p] + yk[(size_t)b * (2 * PDIM) + PDIM + p];
}

extern "C" void kernel_launch(void* const* d_in, const int* in_sizes, int n_in,
                              void* d_out, int out_size, void* d_ws, size_t ws_size,
                              hipStream_t stream) {
    const float* x  = (const float*)d_in[0];
    const float* Wg = (const float*)d_in[1];
    const float* bg = (const float*)d_in[2];
    const float* Wr = (const float*)d_in[3];
    const float* br = (const float*)d_in[4];
    float* out = (float*)d_out;
    char* ws = (char*)d_ws;

    float*  mu     = (float*)(ws + 0);
    float*  sd     = (float*)(ws + 262144);
    float*  gates  = (float*)(ws + 524288);
    int*    topi   = (int*)  (ws + 1048576);
    int*    counts = (int*)  (ws + 1572864);
    int*    lists  = (int*)  (ws + 1572928);
    __bf16* whi    = (__bf16*)(ws + 5242944);
    __bf16* wlo    = (__bf16*)(ws + 6619200);
    float*  w1     = (float*)(ws + 7995456);
    float*  yk     = (float*)(ws + 8000832);

    zero_counts_kernel<<<1, 32, 0, stream>>>(counts);
    const int nW = NEXP * PDIM * LSEQ;
    wprep_kernel<<<(nW + 255) / 256, 256, 0, stream>>>(Wr, whi, wlo, nW);
    w1_kernel<<<(NEXP * PDIM + 255) / 256, 256, 0, stream>>>(Wr, w1);
    gate_kernel<<<BATCH, 256, 0, stream>>>(x, Wg, bg, mu, sd, gates, topi, counts, lists);
    const int totSlots = BATCH * 2 * PDIM;
    init_yk_kernel<<<(totSlots + 255) / 256, 256, 0, stream>>>(x, mu, topi, gates, yk, totSlots);
    dim3 gC(NEXP, BATCH / 16);       // per-expert worst case B rows -> 4096 tiles
    expert_kernel<<<gC, 192, 0, stream>>>(x, whi, wlo, w1, br, mu, sd, gates, counts, lists, yk);
    const int totOut = BATCH * PDIM;
    combine_kernel<<<(totOut + 255) / 256, 256, 0, stream>>>(yk, out, totOut);
}
